// HybridModel_85418309583283
// MI455X (gfx1250) — compile-verified, matrix operand order checked by reading, not run on hardware
//
#include <hip/hip_runtime.h>
#include <hip/hip_bf16.h>
#include <math.h>

typedef __attribute__((ext_vector_type(16))) _Float16     v16h;
typedef __attribute__((ext_vector_type(8)))  float        v8f;
typedef __attribute__((ext_vector_type(4)))  unsigned int v4u;
typedef __attribute__((ext_vector_type(8)))  int          v8i_;
typedef __attribute__((ext_vector_type(4)))  int          v4i_;

#define WMMA_F16(A, Bm, C) \
  __builtin_amdgcn_wmma_f32_16x16x32_f16(false, (A), false, (Bm), (short)0, (C), false, false)

// K index inside a 32-wide chunk for half-slot h (0..15) and lane-group g (0..1).
// ISA 16-bit A-matrix 16x32 layout: lanes 0-15 -> K 0-7,16-23 ; lanes 16-31 -> K 8-15,24-31.
__device__ __forceinline__ int frag_k(int h, int g) {
  return (h & 7) + ((h >> 3) << 4) + (g << 3);
}

__device__ __forceinline__ float softplusf(float v) {
  return (v > 20.f) ? v : log1pf(expf(v));
}

// A-fragment (16 rows x 32 K, f16) for one lane: row fixed, K runs contiguous.
// halves 0..7  = k0 + 8g + 0..7      -> two float4 loads
// halves 8..15 = k0 + 16 + 8g + 0..7 -> two float4 loads
__device__ __forceinline__ v16h load_a_frag(const float* __restrict__ rowptr, int k0, int g) {
  const float4* p = (const float4*)(rowptr + k0 + 8 * g);
  const float4* q = (const float4*)(rowptr + k0 + 16 + 8 * g);
  const float4 x0 = p[0], x1 = p[1], y0 = q[0], y1 = q[1];
  v16h f;
  f[0]  = (_Float16)x0.x; f[1]  = (_Float16)x0.y; f[2]  = (_Float16)x0.z; f[3]  = (_Float16)x0.w;
  f[4]  = (_Float16)x1.x; f[5]  = (_Float16)x1.y; f[6]  = (_Float16)x1.z; f[7]  = (_Float16)x1.w;
  f[8]  = (_Float16)y0.x; f[9]  = (_Float16)y0.y; f[10] = (_Float16)y0.z; f[11] = (_Float16)y0.w;
  f[12] = (_Float16)y1.x; f[13] = (_Float16)y1.y; f[14] = (_Float16)y1.z; f[15] = (_Float16)y1.w;
  return f;
}

// ---------------------------------------------------------------------------
// TDM: issue a 2D tensor-tile load (f32 elements) Global -> LDS.
// D# layout per CDNA5 ISA ch.8: group0 = count/lds_addr/global_addr/type,
// group1 = data_size, tensor dims, tile dims, dim0 stride. Groups 2/3 zero (2D).
// 6-arg builtin form (clang-23 / therock-10.0 toolchain).
// ---------------------------------------------------------------------------
__device__ __forceinline__ void tdm_load_tile_f32_2d(unsigned int lds_off,
                                                     const void* gptr,
                                                     unsigned int tensor_d0,
                                                     unsigned int tensor_d1,
                                                     unsigned int tile_d0,
                                                     unsigned int tile_d1,
                                                     unsigned long long stride0) {
  const unsigned long long ga = (unsigned long long)(uintptr_t)gptr;
  v4u g0;
  g0[0] = 1u;                                             // count=1, user mode
  g0[1] = lds_off;                                        // lds_addr (bytes)
  g0[2] = (unsigned int)(ga & 0xFFFFFFFFu);               // global_addr[31:0]
  g0[3] = (unsigned int)((ga >> 32) & 0x01FFFFFFu)        // global_addr[56:32]
          | 0x80000000u;                                  // type = 2 ("image")
  v8i_ g1;
  g1[0] = 0x00020000;                                     // data_size = 4B
  g1[1] = (int)((tensor_d0 & 0xFFFFu) << 16);             // tensor_dim0[15:0] @ bit48
  g1[2] = (int)((tensor_d0 >> 16) | ((tensor_d1 & 0xFFFFu) << 16));
  g1[3] = (int)((tensor_d1 >> 16) | (tile_d0 << 16));     // tile_dim0 @ bit112
  g1[4] = (int)(tile_d1 & 0xFFFFu);                       // tile_dim1 (tile_dim2=0)
  g1[5] = (int)(unsigned int)(stride0 & 0xFFFFFFFFu);     // tensor_dim0_stride[31:0]
  g1[6] = (int)(unsigned int)((stride0 >> 32) & 0xFFFFu); // stride[47:32]
  g1[7] = 0;
  v4i_ z4 = {};
  v8i_ z8 = {};
  __builtin_amdgcn_tensor_load_to_lds(g0, g1, z4, z4, z8, 0);
}

// ---------------------------------------------------------------------------
// Kernel 1: coefficient chain for the collapsed linear scan.
// d_0 = c (= C_ssm row);  d_{k+1} = A @ d_k;  coeff[T-1-k][m] = B_ssm[m,:] . d_k
// ---------------------------------------------------------------------------
__global__ void ssm_coeff_kernel(const float* __restrict__ A,
                                 const float* __restrict__ Bs,   // [MIN,S]
                                 const float* __restrict__ Cs,   // [1,S]
                                 float* __restrict__ coeff,      // [T,MIN]
                                 int S, int T, int MIN) {
  __shared__ float d0[256];
  __shared__ float d1[256];
  const int tid = threadIdx.x;
  d0[tid] = Cs[tid];
  __syncthreads();
  float* cur = d0;
  float* nxt = d1;
  for (int k = 0; k < T; ++k) {
    if (tid < MIN) {
      float s = 0.f;
      for (int j = 0; j < S; ++j) s += Bs[tid * S + j] * cur[j];
      coeff[(size_t)(T - 1 - k) * MIN + tid] = s;
    }
    float s = 0.f;
    for (int j = 0; j < S; ++j) s += A[(size_t)tid * S + j] * cur[j];
    nxt[tid] = s;
    __syncthreads();
    float* t = cur; cur = nxt; nxt = t;
  }
}

// ---------------------------------------------------------------------------
// Kernel 2: mamba_out[b] = <x_mamba[b,:,:].flatten(), coeff.flatten()>
// Pure HBM stream (512 MB); float4 + prefetch, block-per-row + LDS reduce.
// ---------------------------------------------------------------------------
__global__ void mamba_dot_kernel(const float* __restrict__ xm,
                                 const float* __restrict__ coeff,
                                 float* __restrict__ out, int n) {
  const int b = blockIdx.x;
  const int tid = threadIdx.x;
  const float4* xv = (const float4*)(xm + (size_t)b * n);
  const float4* cv = (const float4*)coeff;
  float s = 0.f;
  const int n4 = n >> 2;
  for (int j = tid; j < n4; j += blockDim.x) {
    if (j + (int)blockDim.x < n4) __builtin_prefetch(&xv[j + blockDim.x], 0, 0);
    const float4 a = xv[j];
    const float4 c = cv[j];
    s += a.x * c.x + a.y * c.y + a.z * c.z + a.w * c.w;
  }
  __shared__ float red[256];
  red[tid] = s;
  __syncthreads();
  for (int off = 128; off > 0; off >>= 1) {
    if (tid < off) red[tid] += red[tid + off];
    __syncthreads();
  }
  if (tid == 0) out[b] = red[0];
}

// ---------------------------------------------------------------------------
// Kernel 3: fused GEMM + bias + ReLU via WMMA f16 (f32 accumulate).
// out[m, u*N + n] = relu( sum_k A[m, u*aColOff + k] * W[u][k][n] + bias[u][n] )
// One wave per block computing a 64x16 C panel (4 WMMAs share one B fragment).
// B panel (32k x 16n f32 tile) staged by the Tensor Data Mover into LDS,
// double-buffered, synchronized with s_wait_tensorcnt.
// ---------------------------------------------------------------------------
__global__ void kan_gemm_relu_kernel(const float* __restrict__ Abase, int lda, int aColOff,
                                     const float* __restrict__ W,     // [U][K][N]
                                     const float* __restrict__ bias,  // [U][N]
                                     float* __restrict__ out, int ldo,
                                     int N, int K) {
  __shared__ float smem[2][32 * 16];   // two 2KB K-tiles of W
  const int u    = blockIdx.z;
  const int m0   = blockIdx.x * 64;
  const int n0   = blockIdx.y * 16;
  const int lane = threadIdx.x;
  const int g    = lane >> 4;
  const int r16  = lane & 15;

  const float* A     = Abase + (size_t)u * aColOff;
  const float* Wu    = W + (size_t)u * K * N;
  const float* biasu = bias + (size_t)u * N;
  // generic pointer to __shared__ carries the wave-relative LDS offset in [31:0]
  const unsigned int lds_base = (unsigned int)(uintptr_t)(void*)&smem[0][0];

  v8f acc[4];
#pragma unroll
  for (int t = 0; t < 4; ++t) acc[t] = (v8f){};

  const int KSTEPS = K / 32;
  tdm_load_tile_f32_2d(lds_base, Wu + (size_t)0 * N + n0,
                       (unsigned)N, (unsigned)K, 16u, 32u, (unsigned long long)N);

  for (int s = 0; s < KSTEPS; ++s) {
    const int k0 = s * 32;
    if (s + 1 < KSTEPS) {
      tdm_load_tile_f32_2d(lds_base + (unsigned)(((s + 1) & 1) * 2048),
                           Wu + (size_t)(k0 + 32) * N + n0,
                           (unsigned)N, (unsigned)K, 16u, 32u, (unsigned long long)N);
      __builtin_amdgcn_s_wait_tensorcnt(1);  // current tile (older) is done
    } else {
      __builtin_amdgcn_s_wait_tensorcnt(0);
    }
    const float* bufp = smem[s & 1];

    v16h bf;                                 // B fragment: n = r16, K mapping per ISA
#pragma unroll
    for (int h = 0; h < 16; ++h) bf[h] = (_Float16)bufp[frag_k(h, g) * 16 + r16];

#pragma unroll
    for (int t = 0; t < 4; ++t) {
      const v16h af = load_a_frag(A + (size_t)(m0 + t * 16 + r16) * lda, k0, g);
      acc[t] = WMMA_F16(af, bf, acc[t]);
    }
  }

  const float bv = biasu[n0 + r16];
#pragma unroll
  for (int t = 0; t < 4; ++t) {
#pragma unroll
    for (int r = 0; r < 8; ++r) {
      const int row = m0 + t * 16 + r + 8 * g;
      float v = acc[t][r] + bv;
      v = v > 0.f ? v : 0.f;
      out[(size_t)row * ldo + (size_t)u * N + n0 + r16] = v;
    }
  }
}

// ---------------------------------------------------------------------------
// Kernel 4: fused KAN inner reduction.
// inner[b,u] = sum_i softplus(h2[b,u,:].ww[u,:,i] + bw[u,i]) * x[b,i]
//            + (h2[b,u,:].wb[u,:,i] + bb[u,i])
// WMMA builds 16x16 tiles of (h2@ww) and (h2@wb); the [B,2,1024] intermediates
// never leave registers. One wave per (16 rows, u).
// ---------------------------------------------------------------------------
__global__ void kan_inner_kernel(const float* __restrict__ h2,   // [B,128] (u block of 64)
                                 const float* __restrict__ x,    // [B,KIN]
                                 const float* __restrict__ ww,   // [U,64,KIN]
                                 const float* __restrict__ wb,   // [U,64,KIN]
                                 const float* __restrict__ bwA,  // [U,KIN]
                                 const float* __restrict__ bbA,  // [U,KIN]
                                 float* __restrict__ inner,      // [B,2]
                                 int KIN) {
  const int u    = blockIdx.z;
  const int b0   = blockIdx.x * 16;
  const int lane = threadIdx.x;
  const int g    = lane >> 4;
  const int r16  = lane & 15;

  // A fragments: h2 rows b0..b0+15, K = 0..63 (two 16x32 chunks), loaded once.
  const float* rowp = h2 + (size_t)(b0 + r16) * 128 + u * 64;
  const v16h a0 = load_a_frag(rowp, 0, g);
  const v16h a1 = load_a_frag(rowp, 32, g);

  const float* wwu = ww + (size_t)u * 64 * KIN;
  const float* wbu = wb + (size_t)u * 64 * KIN;

  float acc[8] = {0.f, 0.f, 0.f, 0.f, 0.f, 0.f, 0.f, 0.f};

  for (int i0 = 0; i0 < KIN; i0 += 16) {
    const int col = i0 + r16;
    v16h w0, w1, q0, q1;
#pragma unroll
    for (int h = 0; h < 16; ++h) {
      const int k = frag_k(h, g);
      w0[h] = (_Float16)wwu[(size_t)k * KIN + col];
      w1[h] = (_Float16)wwu[(size_t)(k + 32) * KIN + col];
      q0[h] = (_Float16)wbu[(size_t)k * KIN + col];
      q1[h] = (_Float16)wbu[(size_t)(k + 32) * KIN + col];
    }
    v8f G = {};
    G = WMMA_F16(a0, w0, G);
    G = WMMA_F16(a1, w1, G);
    v8f Gb = {};
    Gb = WMMA_F16(a0, q0, Gb);
    Gb = WMMA_F16(a1, q1, Gb);

    const float bwv = bwA[(size_t)u * KIN + col];
    const float bbv = bbA[(size_t)u * KIN + col];
#pragma unroll
    for (int r = 0; r < 8; ++r) {
      const int row = b0 + r + 8 * g;
      const float wv = softplusf(G[r] + bwv);
      const float bv = Gb[r] + bbv;
      acc[r] += wv * x[(size_t)row * KIN + col] + bv;
    }
  }

  // cross-lane row reduction: rows 0..7 live in lanes 0..15, rows 8..15 in 16..31
  __shared__ float red[32][8];
#pragma unroll
  for (int r = 0; r < 8; ++r) red[lane][r] = acc[r];
  __syncthreads();
  if (lane < 16) {
    const int row = lane;
    const int gg  = row >> 3;
    const int rr  = row & 7;
    float s = 0.f;
#pragma unroll
    for (int j = 0; j < 16; ++j) s += red[gg * 16 + j][rr];
    inner[(size_t)(b0 + row) * 2 + u] = s;
  }
}

// ---------------------------------------------------------------------------
// Kernel 5: outer generator (input dim UI=2 -> tiny MLP) + final blend.
// ---------------------------------------------------------------------------
__global__ void kan_outer_kernel(const float* __restrict__ inner,  // [B,2]
                                 const float* __restrict__ w1,     // [2,2,128]
                                 const float* __restrict__ b1,     // [2,128]
                                 const float* __restrict__ w2,     // [2,128,64]
                                 const float* __restrict__ b2,     // [2,64]
                                 const float* __restrict__ wwp,    // [2,64,1]
                                 const float* __restrict__ bwp,    // [2,1]
                                 const float* __restrict__ wbp,    // [2,64,1]
                                 const float* __restrict__ bbp,    // [2,1]
                                 const float* __restrict__ mamba_out,
                                 const float* __restrict__ wk,
                                 const float* __restrict__ wm,
                                 float* __restrict__ out) {
  __shared__ float s_w1[512];
  __shared__ float s_b1[256];
  __shared__ float s_b2[128];
  __shared__ float s_ww[128];
  __shared__ float s_wb[128];
  __shared__ float s_misc[4];
  const int tid = threadIdx.x;  // 256
  for (int i = tid; i < 512; i += 256) s_w1[i] = w1[i];
  s_b1[tid] = b1[tid];
  if (tid < 128) { s_b2[tid] = b2[tid]; s_ww[tid] = wwp[tid]; s_wb[tid] = wbp[tid]; }
  if (tid < 2)   { s_misc[tid] = bwp[tid]; s_misc[2 + tid] = bbp[tid]; }
  __syncthreads();

  const int b  = blockIdx.x * 128 + (tid >> 1);
  const int uo = tid & 1;
  const float in0 = inner[(size_t)b * 2 + 0];
  const float in1 = inner[(size_t)b * 2 + 1];

  float h2a[64];
#pragma unroll
  for (int gi = 0; gi < 64; ++gi) h2a[gi] = s_b2[uo * 64 + gi];

  for (int f = 0; f < 128; ++f) {
    float h1f = in0 * s_w1[(uo * 2 + 0) * 128 + f] +
                in1 * s_w1[(uo * 2 + 1) * 128 + f] + s_b1[uo * 128 + f];
    h1f = h1f > 0.f ? h1f : 0.f;
    const float* w2row = w2 + ((size_t)uo * 128 + f) * 64;
#pragma unroll
    for (int gi = 0; gi < 64; ++gi) h2a[gi] += h1f * w2row[gi];
  }

  float wacc = 0.f, bacc = 0.f;
#pragma unroll
  for (int gi = 0; gi < 64; ++gi) {
    const float h2g = h2a[gi] > 0.f ? h2a[gi] : 0.f;
    wacc += h2g * s_ww[uo * 64 + gi];
    bacc += h2g * s_wb[uo * 64 + gi];
  }
  const float weight = softplusf(wacc + s_misc[uo]);
  const float bias   = bacc + s_misc[2 + uo];
  const float outer  = weight * (in0 + in1) + 2.f * bias;

  const float other = __shfl_xor(outer, 1, 32);
  if (uo == 0) {
    const float kan = outer + other;
    out[b] = wk[0] * kan + wm[0] * mamba_out[b];
  }
}

// ---------------------------------------------------------------------------
extern "C" void kernel_launch(void* const* d_in, const int* in_sizes, int n_in,
                              void* d_out, int out_size, void* d_ws, size_t ws_size,
                              hipStream_t stream) {
  const int KIN = 1024, MIND = 64, S = 256, T = 128, H1 = 128, H2 = 64;
  const int B = in_sizes[0] / KIN;  // 16384

  const float* x_kan   = (const float*)d_in[0];
  const float* x_mamba = (const float*)d_in[1];
  const float* in_w1   = (const float*)d_in[2];
  const float* in_b1   = (const float*)d_in[3];
  const float* in_w2   = (const float*)d_in[4];
  const float* in_b2   = (const float*)d_in[5];
  const float* in_ww   = (const float*)d_in[6];
  const float* in_bw   = (const float*)d_in[7];
  const float* in_wb   = (const float*)d_in[8];
  const float* in_bb   = (const float*)d_in[9];
  const float* out_w1  = (const float*)d_in[10];
  const float* out_b1  = (const float*)d_in[11];
  const float* out_w2  = (const float*)d_in[12];
  const float* out_b2  = (const float*)d_in[13];
  const float* out_ww  = (const float*)d_in[14];
  const float* out_bw  = (const float*)d_in[15];
  const float* out_wb  = (const float*)d_in[16];
  const float* out_bb  = (const float*)d_in[17];
  const float* Amat    = (const float*)d_in[18];
  const float* B_ssm   = (const float*)d_in[19];
  const float* C_ssm   = (const float*)d_in[20];
  const float* wk      = (const float*)d_in[21];
  const float* wm      = (const float*)d_in[22];

  float* ws        = (float*)d_ws;
  float* coeff     = ws;                          // T*MIND = 8192
  float* mamba_out = coeff + (size_t)T * MIND;    // B
  float* h1        = mamba_out + B;               // B*256
  float* h2        = h1 + (size_t)B * 256;        // B*128
  float* inner     = h2 + (size_t)B * 128;        // B*2

  // Mamba branch: collapsed linear scan -> coefficient chain + batched dot.
  ssm_coeff_kernel<<<1, 256, 0, stream>>>(Amat, B_ssm, C_ssm, coeff, S, T, MIND);
  mamba_dot_kernel<<<B, 256, 0, stream>>>(x_mamba, coeff, mamba_out, T * MIND);

  // KAN inner generator: two WMMA GEMMs (TDM-staged weights), fused reduction.
  dim3 g1(B / 64, H1 / 16, 2);
  kan_gemm_relu_kernel<<<g1, 32, 0, stream>>>(x_kan, KIN, 0, in_w1, in_b1,
                                              h1, 2 * H1, H1, KIN);
  dim3 g2(B / 64, H2 / 16, 2);
  kan_gemm_relu_kernel<<<g2, 32, 0, stream>>>(h1, 2 * H1, H1, in_w2, in_b2,
                                              h2, 2 * H2, H2, H1);
  dim3 g3(B / 16, 1, 2);
  kan_inner_kernel<<<g3, 32, 0, stream>>>(h2, x_kan, in_ww, in_wb, in_bw, in_bb,
                                          inner, KIN);

  // Outer generator + blend.
  kan_outer_kernel<<<B / 128, 256, 0, stream>>>(inner, out_w1, out_b1, out_w2, out_b2,
                                                out_ww, out_bw, out_wb, out_bb,
                                                mamba_out, wk, wm, (float*)d_out);
}
